// TestModel_59047210385752
// MI455X (gfx1250) — matrix operand order repaired and checked by reading, compiled, not run on hardware
//
#include <hip/hip_runtime.h>
#include <hip/hip_bf16.h>
#include <math.h>

// ---------------------------------------------------------------------------
// Fused NeRF forward for MI455X (gfx1250, wave32, WMMA).
// One workgroup (256 thr = 8 wave32) renders 2 rays (128 sample points).
// All activations live in LDS; GEMMs use v_wmma_f32_16x16x32_f16.
// N-partitioned GEMMs: each wave owns N-columns and holds its B-fragments in
// VGPRs across all M-tiles -> 8x less L2 weight traffic than M-partitioning.
// ---------------------------------------------------------------------------

#define HH 100
#define WW 100
#define FX 86.6f
#define FY 86.6f
#define NEARP 2.0f
#define FARP 6.0f
#define DINF 10000000000.0f
#define NSAMP 64

typedef __attribute__((ext_vector_type(16))) _Float16 v16h;
typedef __attribute__((ext_vector_type(8)))  _Float16 v8h;
typedef __attribute__((ext_vector_type(8)))  float    v8f;

// Weight regions in d_ws (f16), padded for WMMA
#define W1_ELEMS (64*256)     // 63->64 rows
#define W2_ELEMS (256*256)
#define WD_ELEMS (288*128)    // 283->288 rows
#define WS_ELEMS (256)
#define WR_ELEMS (384)        // 128x3
#define WTOT (W1_ELEMS+W2_ELEMS+WD_ELEMS+WS_ELEMS+WR_ELEMS)

// LDS layout (bytes)
#define OFF_ENC   0            // 128 x 64 f16   (16384)
#define OFF_DIRB  16384        // 128 x 32 f16   ( 8192)
#define OFF_H1    24576        // 128 x 256 f16  (65536)  -- reused for hd 128x128
#define OFF_H2    90112        // 128 x 256 f16  (65536)
#define OFF_SIG   155648       // 128 f32        (  512)
#define OFF_RGB   156160       // 128 x 3 f32    ( 1536)
#define OFF_Z     157696       // 64 f32         (  256)
#define OFF_DIST  157952       // 64 f32         (  256)
#define OFF_RAY   158208       // rays_d[2][3], norm[2], dirE[2][27] f32 (248)
#define SMEM_BYTES 158464

// --------------------------- weight prep (f32 -> f16) ----------------------
__global__ void prep_weights(const float* __restrict__ w1,
                             const float* __restrict__ w2,
                             const float* __restrict__ wdir,
                             const float* __restrict__ wsig,
                             const float* __restrict__ wrgb,
                             _Float16* __restrict__ W1,
                             _Float16* __restrict__ W2,
                             _Float16* __restrict__ WD,
                             _Float16* __restrict__ WS,
                             _Float16* __restrict__ WR) {
  int idx = blockIdx.x * 256 + threadIdx.x;
  if (idx < W1_ELEMS) {
    int k = idx >> 8, n = idx & 255;
    W1[idx] = (k < 63) ? (_Float16)w1[k * 256 + n] : (_Float16)0.0f;
  }
  int i2 = idx - W1_ELEMS;
  if (i2 >= 0 && i2 < W2_ELEMS) W2[i2] = (_Float16)w2[i2];
  int i3 = idx - (W1_ELEMS + W2_ELEMS);
  if (i3 >= 0 && i3 < WD_ELEMS) {
    int k = i3 >> 7, n = i3 & 127;
    WD[i3] = (k < 283) ? (_Float16)wdir[k * 128 + n] : (_Float16)0.0f;
  }
  int i4 = idx - (W1_ELEMS + W2_ELEMS + WD_ELEMS);
  if (i4 >= 0 && i4 < WS_ELEMS) WS[i4] = (_Float16)wsig[i4];
  int i5 = idx - (W1_ELEMS + W2_ELEMS + WD_ELEMS + WS_ELEMS);
  if (i5 >= 0 && i5 < WR_ELEMS) WR[i5] = (_Float16)wrgb[i5];
}

// --------------------------- fragment helpers ------------------------------
// A (16x32 f16, MxK): lane<16 -> M=lane, K = kbase + e%8 + (e/8)*16 (hi=0)
//                     lane>=16 adds +8 to K. Two 16B LDS chunks per lane.
__device__ __forceinline__ v16h load_a(const _Float16* __restrict__ src,
                                       int stride, int m0, int kbase, int lane) {
  int hi  = (lane >> 4) & 1;
  int row = lane & 15;
  const _Float16* p = src + (m0 + row) * stride + kbase + hi * 8;
  v8h lo  = *(const v8h*)(p);
  v8h hi8 = *(const v8h*)(p + 16);
  v16h a;
#pragma unroll
  for (int e = 0; e < 8; ++e) { a[e] = lo[e]; a[e + 8] = hi8[e]; }
  return a;
}

// B (32x16 f16, KxN): N = lane&15, K = kbase + e%8 + (e/8)*16 + 8*(lane>=16).
__device__ __forceinline__ v16h load_b(const _Float16* __restrict__ w,
                                       int N, int kbase, int nt, int lane) {
  int hi = (lane >> 4) & 1;
  const _Float16* p = w + (kbase + hi * 8) * N + nt * 16 + (lane & 15);
  v16h b;
#pragma unroll
  for (int e = 0; e < 8; ++e) { b[e] = p[e * N]; b[e + 8] = p[(e + 16) * N]; }
  return b;
}

// C/D (16x16 f32): VGPR r, lane -> M = r + 8*(lane>=16), N = lane&15.
__device__ __forceinline__ void store_cd(_Float16* __restrict__ dst, int stride,
                                         int m0, int nt, int lane, v8f acc,
                                         float bv) {
  int hi  = (lane >> 4) & 1;
  int n   = nt * 16 + (lane & 15);
#pragma unroll
  for (int r = 0; r < 8; ++r) {
    float v = fmaxf(acc[r] + bv, 0.0f);   // fused bias + ReLU
    dst[(m0 + r + 8 * hi) * stride + n] = (_Float16)v;
  }
}

// --------------------------- fused NeRF kernel -----------------------------
__global__ void __launch_bounds__(256)
nerf_fused(const float* __restrict__ c2w, const float* __restrict__ t_rand,
           const float* __restrict__ b1, const float* __restrict__ b2,
           const float* __restrict__ bsig, const float* __restrict__ bdir,
           const float* __restrict__ brgb,
           const _Float16* __restrict__ W1, const _Float16* __restrict__ W2,
           const _Float16* __restrict__ WD, const _Float16* __restrict__ WS,
           const _Float16* __restrict__ WR,
           float* __restrict__ out) {
  extern __shared__ char smem[];
  _Float16* encS  = (_Float16*)(smem + OFF_ENC);   // 128 x 64
  _Float16* dirbS = (_Float16*)(smem + OFF_DIRB);  // 128 x 32
  _Float16* h1S   = (_Float16*)(smem + OFF_H1);    // 128 x 256
  _Float16* h2S   = (_Float16*)(smem + OFF_H2);    // 128 x 256
  _Float16* hdS   = (_Float16*)(smem + OFF_H1);    // 128 x 128 (reuse h1)
  float*    sigS  = (float*)(smem + OFF_SIG);      // 128
  float*    rgbS  = (float*)(smem + OFF_RGB);      // 128 x 3
  float*    zS    = (float*)(smem + OFF_Z);        // 64
  float*    distS = (float*)(smem + OFF_DIST);     // 64
  float*    raydS = (float*)(smem + OFF_RAY);      // [2][3]
  float*    normS = raydS + 6;                     // [2]
  float*    dirES = normS + 2;                     // [2][27]

  const int t    = threadIdx.x;
  const int wave = t >> 5;
  const int lane = t & 31;
  const int r0   = blockIdx.x * 2;

  // ---- Stage 0a: per-ray dirs + dir posenc; stratified z samples ----
  if (t < 2) {
    int r = r0 + t;
    float fi = (float)(r % WW), fj = (float)(r / WW);
    float dx = (fi - 0.5f * WW) / FX;
    float dy = -(fj - 0.5f * HH) / FY;
    float dz = -1.0f;
    float rd[3];
#pragma unroll
    for (int c = 0; c < 3; ++c)
      rd[c] = dx * c2w[c * 4 + 0] + dy * c2w[c * 4 + 1] + dz * c2w[c * 4 + 2];
    float nrm = sqrtf(rd[0] * rd[0] + rd[1] * rd[1] + rd[2] * rd[2]);
    normS[t] = nrm;
#pragma unroll
    for (int c = 0; c < 3; ++c) raydS[t * 3 + c] = rd[c];
    float vd[3] = {rd[0] / nrm, rd[1] / nrm, rd[2] / nrm};
#pragma unroll
    for (int c = 0; c < 3; ++c) dirES[t * 27 + c] = vd[c];
#pragma unroll
    for (int l = 0; l < 4; ++l) {
      float f = (float)(1 << l);
#pragma unroll
      for (int c = 0; c < 3; ++c) {
        dirES[t * 27 + 3 + 6 * l + c]     = sinf(vd[c] * f);
        dirES[t * 27 + 3 + 6 * l + 3 + c] = cosf(vd[c] * f);
      }
    }
  }
  if (t < NSAMP) {
    auto zlin = [](int s) {
      float tv = (float)s / (float)(NSAMP - 1);
      return NEARP * (1.0f - tv) + FARP * tv;
    };
    float lower = (t == 0) ? zlin(0) : 0.5f * (zlin(t - 1) + zlin(t));
    float upper = (t == NSAMP - 1) ? zlin(NSAMP - 1) : 0.5f * (zlin(t) + zlin(t + 1));
    zS[t] = lower + (upper - lower) * t_rand[t];
  }
  __syncthreads();

  // ---- Stage 0b: dists; point posenc into LDS (f16) ----
  if (t < NSAMP) distS[t] = (t < NSAMP - 1) ? (zS[t + 1] - zS[t]) : DINF;
  if (t < 128) {
    int ray = t >> 6, s = t & 63;
    float zv = zS[s];
    float p[3];
#pragma unroll
    for (int c = 0; c < 3; ++c)
      p[c] = c2w[c * 4 + 3] + raydS[ray * 3 + c] * zv;
    _Float16* e = encS + t * 64;
#pragma unroll
    for (int c = 0; c < 3; ++c) e[c] = (_Float16)p[c];
#pragma unroll
    for (int l = 0; l < 10; ++l) {
      float f = (float)(1 << l);
#pragma unroll
      for (int c = 0; c < 3; ++c) {
        e[3 + 6 * l + c]     = (_Float16)sinf(p[c] * f);
        e[3 + 6 * l + 3 + c] = (_Float16)cosf(p[c] * f);
      }
    }
    e[63] = (_Float16)0.0f;
    _Float16* db = dirbS + t * 32;
#pragma unroll
    for (int jj = 0; jj < 32; ++jj)
      db[jj] = (jj < 27) ? (_Float16)dirES[ray * 27 + jj] : (_Float16)0.0f;
  }
  __syncthreads();

  // ---- Stage 1: h1 = relu(enc @ W1 + b1)   M=128 K=64 N=256 ----
  // N-partitioned: wave owns nt = {wave, wave+8}; B held in VGPRs over M-tiles.
  for (int half = 0; half < 2; ++half) {
    const int nt = wave + half * 8;
    v16h breg[2];
#pragma unroll
    for (int kt = 0; kt < 2; ++kt) breg[kt] = load_b(W1, 256, kt * 32, nt, lane);
    const float bv = b1[nt * 16 + (lane & 15)];
    for (int mt = 0; mt < 8; ++mt) {
      v8f acc = {};
#pragma unroll
      for (int kt = 0; kt < 2; ++kt) {
        v16h a = load_a(encS, 64, mt * 16, kt * 32, lane);
        acc = __builtin_amdgcn_wmma_f32_16x16x32_f16(false, a, false, breg[kt],
                                                     (short)0, acc, false, false);
      }
      store_cd(h1S, 256, mt * 16, nt, lane, acc, bv);
    }
  }
  __syncthreads();

  // ---- Stage 2: h2 = relu(h1 @ W2 + b2)   M=128 K=256 N=256 ----
  for (int half = 0; half < 2; ++half) {
    const int nt = wave + half * 8;
    v16h breg[8];
#pragma unroll
    for (int kt = 0; kt < 8; ++kt) breg[kt] = load_b(W2, 256, kt * 32, nt, lane);
    const float bv = b2[nt * 16 + (lane & 15)];
    for (int mt = 0; mt < 8; ++mt) {
      v8f acc = {};
#pragma unroll
      for (int kt = 0; kt < 8; ++kt) {
        v16h a = load_a(h1S, 256, mt * 16, kt * 32, lane);
        acc = __builtin_amdgcn_wmma_f32_16x16x32_f16(false, a, false, breg[kt],
                                                     (short)0, acc, false, false);
      }
      store_cd(h2S, 256, mt * 16, nt, lane, acc, bv);
    }
  }
  __syncthreads();

  // ---- Stage 3: sigma head (N=1, VALU dot) ----
  if (t < 128) {
    float s = bsig[0];
    const _Float16* hrow = h2S + t * 256;
#pragma unroll 8
    for (int k = 0; k < 256; ++k) s += (float)hrow[k] * (float)WS[k];
    sigS[t] = s;
  }

  // ---- Stage 4: hd = relu([h2, enc_dir] @ WD + bdir)  M=128 K=288 N=128 ----
  {
    const int nt = wave;          // one n-tile per wave (N = 128)
    v16h breg[9];
#pragma unroll
    for (int kt = 0; kt < 9; ++kt) breg[kt] = load_b(WD, 128, kt * 32, nt, lane);
    const float bv = bdir[nt * 16 + (lane & 15)];
    for (int mt = 0; mt < 8; ++mt) {
      v8f acc = {};
#pragma unroll
      for (int kt = 0; kt < 9; ++kt) {
        v16h a = (kt < 8) ? load_a(h2S, 256, mt * 16, kt * 32, lane)
                          : load_a(dirbS, 32, mt * 16, 0, lane);
        acc = __builtin_amdgcn_wmma_f32_16x16x32_f16(false, a, false, breg[kt],
                                                     (short)0, acc, false, false);
      }
      store_cd(hdS, 128, mt * 16, nt, lane, acc, bv);
    }
  }
  __syncthreads();

  // ---- Stage 5: rgb head (N=3, VALU dot + sigmoid) ----
  if (t < 128) {
    const _Float16* hrow = hdS + t * 128;
#pragma unroll
    for (int c = 0; c < 3; ++c) {
      float s = brgb[c];
#pragma unroll 8
      for (int k = 0; k < 128; ++k) s += (float)hrow[k] * (float)WR[k * 3 + c];
      rgbS[t * 3 + c] = 1.0f / (1.0f + expf(-s));
    }
  }
  __syncthreads();

  // ---- Stage 6: volume render (2 rays, sequential cumprod) ----
  if (t < 2) {
    float T = 1.0f, a0 = 0.0f, a1 = 0.0f, a2 = 0.0f;
    float nrm = normS[t];
    for (int s = 0; s < NSAMP; ++s) {
      int m = t * NSAMP + s;
      float alpha = 1.0f - expf(-fmaxf(sigS[m], 0.0f) * distS[s] * nrm);
      float w = alpha * T;
      a0 += w * rgbS[m * 3 + 0];
      a1 += w * rgbS[m * 3 + 1];
      a2 += w * rgbS[m * 3 + 2];
      T *= (1.0f - alpha);
    }
    int r = r0 + t;
    out[r * 3 + 0] = a0;
    out[r * 3 + 1] = a1;
    out[r * 3 + 2] = a2;
  }
}

// --------------------------- launcher --------------------------------------
extern "C" void kernel_launch(void* const* d_in, const int* in_sizes, int n_in,
                              void* d_out, int out_size, void* d_ws, size_t ws_size,
                              hipStream_t stream) {
  const float* c2w    = (const float*)d_in[0];
  const float* t_rand = (const float*)d_in[1];
  const float* w1     = (const float*)d_in[2];
  const float* b1     = (const float*)d_in[3];
  const float* w2     = (const float*)d_in[4];
  const float* b2     = (const float*)d_in[5];
  const float* w_sig  = (const float*)d_in[6];
  const float* b_sig  = (const float*)d_in[7];
  const float* w_dir  = (const float*)d_in[8];
  const float* b_dir  = (const float*)d_in[9];
  const float* w_rgb  = (const float*)d_in[10];
  const float* b_rgb  = (const float*)d_in[11];

  _Float16* W1 = (_Float16*)d_ws;
  _Float16* W2 = W1 + W1_ELEMS;
  _Float16* WD = W2 + W2_ELEMS;
  _Float16* WS = WD + WD_ELEMS;
  _Float16* WR = WS + WS_ELEMS;

  (void)hipFuncSetAttribute((const void*)nerf_fused,
                            hipFuncAttributeMaxDynamicSharedMemorySize,
                            SMEM_BYTES);

  prep_weights<<<(WTOT + 255) / 256, 256, 0, stream>>>(
      w1, w2, w_dir, w_sig, w_rgb, W1, W2, WD, WS, WR);

  const int nblocks = (HH * WW) / 2;  // 2 rays per workgroup
  nerf_fused<<<nblocks, 256, SMEM_BYTES, stream>>>(
      c2w, t_rand, b1, b2, b_sig, b_dir, b_rgb, W1, W2, WD, WS, WR,
      (float*)d_out);
}
